// Proto_contrast_loss_65781719105993
// MI455X (gfx1250) — matrix operand order repaired and checked by reading, not compiled, not run on hardware
//
#include <hip/hip_runtime.h>
#include <hip/hip_bf16.h>
#include <math.h>

// ---------------- problem constants (fixed shapes from the reference) -------
enum : int {
  BB  = 2,      // batch
  EE  = 256,    // embedding dim (K of the big GEMM)
  HWH = 1024,   // H*W
  CC  = 80,     // classes
  QQ  = 1024,   // queue size
};
#define TEMP_INV (1.0f / 0.07f)
#define LOSS_W   0.01f

// LDS staging geometry for the main kernel (dynamic LDS; 320KB/WGP available)
#define NQ     8                 // queue rows per staged chunk
#define LROW   264               // padded row stride in halves (256 data + 8 pad)
#define LQTILE (16 * LROW)       // one queue row's 16-class tile in LDS (halves)
// double buffer: 2 * NQ * LQTILE * 2B = 135,168 bytes of dynamic LDS

typedef __attribute__((ext_vector_type(16))) _Float16 v16h;
typedef __attribute__((ext_vector_type(8)))  float    v8f;
typedef unsigned int u32x4 __attribute__((ext_vector_type(4)));
typedef int          i32x4 __attribute__((ext_vector_type(4)));
typedef int          i32x8 __attribute__((ext_vector_type(8)));

union Frag16 {
  v16h  h;
  uint4 u[2];
};

// ---- Tensor Data Mover availability / arity detection ----------------------
#if defined(__has_builtin)
#if __has_builtin(__builtin_amdgcn_tensor_load_to_lds)
#define USE_TDM 1
#if __has_include(<hip/amd_detail/amd_gfx1250_TDM.h>)
#define TDM_ARGS6 1   // amdgpu-toolchain (therock headers) => 6-arg builtin
#endif
#endif
#endif

// ---------------------------------------------------------------------------
// K0: softmax over C with temperature 0.001, stored TRANSPOSED as (B, C, HW).
// ---------------------------------------------------------------------------
__global__ __launch_bounds__(256) void k_softmax_T(const float* __restrict__ score,
                                                   float* __restrict__ smT) {
  int idx = blockIdx.x * blockDim.x + threadIdx.x;   // over B*HW
  if (idx >= BB * HWH) return;
  int b = idx / HWH, h = idx % HWH;
  const float* sp = score + (size_t)b * CC * HWH + h;
  float mx = -3.0e38f;
  for (int c = 0; c < CC; ++c) mx = fmaxf(mx, sp[(size_t)c * HWH]);
  float sum = 0.f;
  for (int c = 0; c < CC; ++c) sum += __expf((sp[(size_t)c * HWH] - mx) * 1000.0f);
  float rs = 1.0f / sum;
  float* op = smT + (size_t)b * CC * HWH + h;
  for (int c = 0; c < CC; ++c)
    op[(size_t)c * HWH] = __expf((sp[(size_t)c * HWH] - mx) * 1000.0f) * rs;
}

// ---------------------------------------------------------------------------
// K1: queue (Q,E,C) f32 -> queue_t (Q,C,E) f16 for rows q >= B.
// ---------------------------------------------------------------------------
__global__ __launch_bounds__(256) void k_queue_t(const float* __restrict__ queue,
                                                 _Float16* __restrict__ qt) {
  size_t total = (size_t)(QQ - BB) * CC * EE;
  size_t idx = (size_t)blockIdx.x * blockDim.x + threadIdx.x;
  if (idx >= total) return;
  int e = (int)(idx % EE);
  size_t t = idx / EE;
  int c = (int)(t % CC);
  int q = (int)(t / CC) + BB;
  qt[idx + (size_t)BB * CC * EE] =
      (_Float16)queue[((size_t)q * EE + e) * CC + c];
}

// ---------------------------------------------------------------------------
// K2: prototypes (cate_protos -> queue_t rows 0..B-1, pos_proto -> pospT).
// ---------------------------------------------------------------------------
__global__ __launch_bounds__(256) void k_protos(const float* __restrict__ feat,
                                                const float* __restrict__ smT,
                                                const float* __restrict__ center,
                                                _Float16* __restrict__ qt,
                                                _Float16* __restrict__ pospT) {
  int b = blockIdx.x / CC;
  int c = blockIdx.x % CC;
  int e = threadIdx.x;
  const float* fp = feat   + ((size_t)b * EE + e) * HWH;
  const float* sp = smT    + ((size_t)b * CC + c) * HWH;
  const float* kp = center + ((size_t)b * CC + c) * HWH;
  float accP = 0.f, accK = 0.f;
  for (int h = 0; h < HWH; ++h) {
    float f = fp[h];
    accP += f * sp[h];
    accK += f * kp[h];
  }
  __shared__ float red[256];
  red[e] = accP * accP; __syncthreads();
  for (int s = 128; s > 0; s >>= 1) {
    if (e < s) red[e] += red[e + s];
    __syncthreads();
  }
  float sumP2 = red[0];
  __syncthreads();
  red[e] = accK * accK; __syncthreads();
  for (int s = 128; s > 0; s >>= 1) {
    if (e < s) red[e] += red[e + s];
    __syncthreads();
  }
  float sumK2 = red[0];
  float rP = 1.0f / fmaxf(sqrtf(sumP2), 1e-5f);
  float rK = 1.0f / fmaxf(sqrtf(sumK2), 1e-5f);
  qt   [((size_t)b * CC + c) * EE + e] = (_Float16)(accP * rP);
  pospT[((size_t)b * CC + c) * EE + e] = (_Float16)(accK * rK);
}

// ---------------------------------------------------------------------------
// K3: feat_n transposed: (B,E,HW) f32 -> (B,HW,E) f16, L2-normalized over E.
// ---------------------------------------------------------------------------
__global__ __launch_bounds__(256) void k_featn_t(const float* __restrict__ feat,
                                                 _Float16* __restrict__ ft) {
  int b  = blockIdx.x / (HWH / 32);
  int h0 = (blockIdx.x % (HWH / 32)) * 32;
  int tid = threadIdx.x;
  int hl = tid & 31;
  int eg = tid >> 5;                 // 0..7
  float ss = 0.f;
  for (int e = eg; e < EE; e += 8) {
    float f = feat[((size_t)b * EE + e) * HWH + h0 + hl];
    ss += f * f;
  }
  __shared__ float part[8][32];
  part[eg][hl] = ss;
  __syncthreads();
  __shared__ float rnorm[32];
  if (tid < 32) {
    float t = 0.f;
    for (int g = 0; g < 8; ++g) t += part[g][tid];
    rnorm[tid] = 1.0f / fmaxf(sqrtf(t), 1e-5f);
  }
  __syncthreads();
  for (int h = 0; h < 32; ++h) {
    float f = feat[((size_t)b * EE + tid) * HWH + h0 + h];
    ft[((size_t)b * HWH + h0 + h) * EE + tid] = (_Float16)(f * rnorm[h]);
  }
}

// ---------------------------------------------------------------------------
// TDM staging of one queue chunk (NQ rows x 16 classes x 256 halves) into LDS.
// 2D D# per ISA 8.3/8.4: dim0 = 4096 contiguous halves (16 classes * 256),
// dim1 = NQ rows with stride C*E halves. pad_enable inserts 4 dwords after
// every 128 dwords stored -> LDS rows padded to LROW halves (bank spread).
// ---------------------------------------------------------------------------
__device__ __forceinline__ void tdm_stage(const _Float16* gsrc, _Float16* ldst) {
#if defined(USE_TDM)
  size_t   ga    = (size_t)(const void*)gsrc;
  uint32_t laddr = (uint32_t)(size_t)(void*)ldst;  // low 32 bits = LDS offset
  u32x4 g0 = { 1u,                                     // count=1, user mode
               laddr,                                  // lds_addr
               (uint32_t)ga,                           // global_addr[31:0]
               (uint32_t)((ga >> 32) & 0x1FFFFFFu) |   // global_addr[56:32]
                   (2u << 30) };                       // type=2 ("image")
  i32x8 g1 = { (int)((1u << 16) |                      // data_size = 2 bytes
                     (1u << 20) |                      // pad_enable
                     (6u << 22) |                      // pad_interval: 128 dw
                     (3u << 25)),                      // pad_amount: 4 dw
               (int)(4096u << 16),                     // tensor_dim0[15:0]
               (int)((unsigned)NQ << 16),              // tensor_dim1[15:0]
               (int)(4096u << 16),                     // tile_dim0 = 4096
               (int)NQ,                                // tile_dim1 = NQ
               (int)(CC * EE),                         // tensor_dim0_stride
               0, 0 };
  i32x4 z4 = {0, 0, 0, 0};
#if defined(TDM_ARGS6)
  i32x8 z8 = {0, 0, 0, 0, 0, 0, 0, 0};
  __builtin_amdgcn_tensor_load_to_lds(g0, g1, z4, z4, z8, 0);
#else
  __builtin_amdgcn_tensor_load_to_lds(g0, g1, z4, z4, 0);
#endif
#else
  (void)gsrc; (void)ldst;
#endif
}

// Fallback: cooperative global->LDS copy with manual row padding.
__device__ __forceinline__ void coop_copy(const _Float16* gsrc, _Float16* ldst) {
  for (int i = threadIdx.x; i < NQ * 16 * 32; i += 256) {   // 16B chunks
    int ql  = i / 512;
    int rem = i % 512;
    int c   = rem >> 5;
    int w   = rem & 31;
    const uint4* s = (const uint4*)(gsrc + ((size_t)ql * CC + c) * EE) + w;
    uint4* d = (uint4*)(ldst + (size_t)ql * LQTILE + (size_t)c * LROW) + w;
    *d = *s;
  }
}

// ---------------------------------------------------------------------------
// K4: fused big GEMM + streaming sum-exp over Q, TDM double-buffered.
// 8 waves/block, each wave one 16-row M-tile (block = 128 HW rows).
// Grid: x = HW/128 (=8), y = C/16 (=5), z = B (=2).
// WMMA f16 k order used by this code (contiguous, not the ISA table order; wave32):
//   A 16x32: lane&15=M, 16 contiguous K at koff=(lane>>4)*16
//   B 32x16: lane&15=N, 16 contiguous K at koff=(lane>>4)*16
//
// Numerics: feat_n is unit-norm; queue rows have norm 0.1 (q>=B) or 1.0
// (proto rows), so |logit| <= 1/0.07 ~= 14.3 and sum_q exp(logit) <= 3.3e6
// -- safely inside f32 range. A plain running sum of exp with one final log
// is therefore exact enough and halves the transcendental work per q.
// ---------------------------------------------------------------------------
__global__ __launch_bounds__(256) void k_main(const _Float16* __restrict__ ft,
                                              const _Float16* __restrict__ qt,
                                              const _Float16* __restrict__ pospT,
                                              float* __restrict__ s_out,
                                              float* __restrict__ pos_out,
                                              float* __restrict__ neg0_out) {
  extern __shared__ _Float16 lds_b[];          // [2][NQ * LQTILE], dynamic

  const int b    = blockIdx.z;
  const int c0   = blockIdx.y * 16;
  const int wave = threadIdx.x >> 5;
  const int lane = threadIdx.x & 31;
  const int m0   = blockIdx.x * 128 + wave * 16;
  const int n    = lane & 15;
  const int hi   = lane >> 4;

  // ---- 8 A fragments held in registers ----
  const _Float16* arow = ft + ((size_t)b * HWH + m0 + n) * EE;
  Frag16 A[8];
#pragma unroll
  for (int kt = 0; kt < 8; ++kt) {
    int off = kt * 32 + hi * 16;
    A[kt].u[0] = *(const uint4*)(arow + off);
    A[kt].u[1] = *(const uint4*)(arow + off + 8);
  }

  // ---- pos GEMM (tiny; straight from global) ----
  {
    const _Float16* prow = pospT + ((size_t)b * CC + c0 + n) * EE;
    v8f accp = {};
#pragma unroll
    for (int kt = 0; kt < 8; ++kt) {
      Frag16 Bf;
      int off = kt * 32 + hi * 16;
      Bf.u[0] = *(const uint4*)(prow + off);
      Bf.u[1] = *(const uint4*)(prow + off + 8);
      accp = __builtin_amdgcn_wmma_f32_16x16x32_f16(
          false, A[kt].h, false, Bf.h, (short)0, accp, false, false);
    }
#pragma unroll
    for (int j = 0; j < 8; ++j) {
      int m = m0 + j + hi * 8;
      pos_out[((size_t)b * HWH + m) * CC + c0 + n] = accp[j] * TEMP_INV;
    }
  }

  // ---- queue chunks: TDM (or cooperative) double-buffered staging ----
  const _Float16* qbase = qt + (size_t)c0 * EE;          // q = 0, this c-tile
  const size_t chunk_stride = (size_t)NQ * CC * EE;

#if defined(USE_TDM)
  if (threadIdx.x < 32) tdm_stage(qbase, &lds_b[0]);
  __builtin_amdgcn_s_wait_tensorcnt(0);
#else
  coop_copy(qbase, &lds_b[0]);
#endif
  __syncthreads();

  float rsum[8];
#pragma unroll
  for (int j = 0; j < 8; ++j) rsum[j] = 0.f;

  const int NCHUNK = QQ / NQ;
  for (int chunk = 0; chunk < NCHUNK; ++chunk) {
    _Float16* bufp = lds_b + (size_t)(chunk & 1) * (NQ * LQTILE);
    if (chunk + 1 < NCHUNK) {
      _Float16* nxtp = lds_b + (size_t)((chunk + 1) & 1) * (NQ * LQTILE);
#if defined(USE_TDM)
      if (threadIdx.x < 32)
        tdm_stage(qbase + (size_t)(chunk + 1) * chunk_stride, nxtp);
#else
      coop_copy(qbase + (size_t)(chunk + 1) * chunk_stride, nxtp);
#endif
    }

#pragma unroll
    for (int ql = 0; ql < NQ; ++ql) {
      const _Float16* brow = bufp + (size_t)ql * LQTILE + (size_t)n * LROW;
      v8f acc = {};
#pragma unroll
      for (int kt = 0; kt < 8; ++kt) {
        Frag16 Bf;                               // -> ds_load_b128 x2
        int off = kt * 32 + hi * 16;
        Bf.u[0] = *(const uint4*)(brow + off);
        Bf.u[1] = *(const uint4*)(brow + off + 8);
        acc = __builtin_amdgcn_wmma_f32_16x16x32_f16(
            false, A[kt].h, false, Bf.h, (short)0, acc, false, false);
      }
      if (chunk == 0 && ql == 0) {   // save q==0 slice (target logits, cls>=1)
#pragma unroll
        for (int j = 0; j < 8; ++j) {
          int m = m0 + j + hi * 8;
          neg0_out[((size_t)b * HWH + m) * CC + c0 + n] = acc[j] * TEMP_INV;
        }
      }
#pragma unroll
      for (int j = 0; j < 8; ++j)
        rsum[j] += __expf(acc[j] * TEMP_INV);    // bounded: no overflow (see hdr)
    }

#if defined(USE_TDM)
    __builtin_amdgcn_s_wait_tensorcnt(0);   // next chunk landed in LDS
#endif
    __syncthreads();                        // all waves done with this buffer
  }

#pragma unroll
  for (int j = 0; j < 8; ++j) {
    int m = m0 + j + hi * 8;
    s_out[((size_t)b * HWH + m) * CC + c0 + n] = __logf(rsum[j]);
  }
}

// ---------------------------------------------------------------------------
// K5a: zero per-class accumulators.
// ---------------------------------------------------------------------------
__global__ void k_zero(float* __restrict__ p, int nfloats) {
  int i = blockIdx.x * blockDim.x + threadIdx.x;
  if (i < nfloats) p[i] = 0.f;
}

// ---------------------------------------------------------------------------
// K5: per-pixel masked CE accumulated per class.
// ---------------------------------------------------------------------------
__global__ __launch_bounds__(256) void k_loss_pix(const float* __restrict__ s,
                                                  const float* __restrict__ pos,
                                                  const float* __restrict__ neg0,
                                                  const float* __restrict__ center,
                                                  float* __restrict__ sums,
                                                  float* __restrict__ cnts) {
  int idx = blockIdx.x * blockDim.x + threadIdx.x;  // over B*HW
  if (idx >= BB * HWH) return;
  int b = idx / HWH, h = idx % HWH;
  const float* sp = s    + ((size_t)b * HWH + h) * CC;
  const float* pp = pos  + ((size_t)b * HWH + h) * CC;
  const float* np = neg0 + ((size_t)b * HWH + h) * CC;

  float M = -3.0e38f;
  for (int c = 0; c < CC; ++c) M = fmaxf(M, sp[c]);
  float T = 0.f;
  for (int c = 0; c < CC; ++c) T += __expf(sp[c] - M);

  for (int cls = 0; cls < CC; ++cls) {
    float cmv = center[((size_t)b * CC + cls) * HWH + h];
    if (cmv > 0.f) {
      float Tex  = T - __expf(sp[cls] - M);
      float lneg = M + __logf(fmaxf(Tex, 1e-30f));
      float p    = pp[cls];
      float a    = fmaxf(p, lneg);
      float bm   = fminf(p, lneg);
      float lt   = a + log1pf(__expf(bm - a));
      float tgt  = (cls == 0) ? pp[0] : np[cls - 1];
      atomicAdd(&sums[cls], lt - tgt);
      atomicAdd(&cnts[cls], 1.0f);
    }
  }
}

// ---------------------------------------------------------------------------
// K6: final scalar.
// ---------------------------------------------------------------------------
__global__ __launch_bounds__(128) void k_final(const float* __restrict__ sums,
                                               const float* __restrict__ cnts,
                                               float* __restrict__ out) {
  __shared__ float red[128];
  int t = threadIdx.x;
  float v = 0.f;
  if (t < CC) {
    float cn = cnts[t];
    if (cn > 0.f) v = sums[t] / fmaxf(cn, 1.0f);
  }
  red[t] = v;
  __syncthreads();
  for (int s = 64; s > 0; s >>= 1) {
    if (t < s) red[t] += red[t + s];
    __syncthreads();
  }
  if (t == 0) out[0] = LOSS_W * red[0];
}

// ---------------------------------------------------------------------------
extern "C" void kernel_launch(void* const* d_in, const int* in_sizes, int n_in,
                              void* d_out, int out_size, void* d_ws, size_t ws_size,
                              hipStream_t stream) {
  const float* feature = (const float*)d_in[0];  // (B,E,HW)
  const float* center  = (const float*)d_in[1];  // (B,C,HW)
  const float* score   = (const float*)d_in[2];  // (B,C,HW)
  const float* queue   = (const float*)d_in[3];  // (Q,E,C)
  float* out = (float*)d_out;

  char* ws = (char*)d_ws;
  size_t off = 0;
  auto carve = [&](size_t bytes) -> char* {
    char* p = ws + off;
    off = (off + bytes + 255) & ~(size_t)255;
    return p;
  };
  float*    smT   = (float*)   carve((size_t)BB * CC * HWH * 4);
  _Float16* qt    = (_Float16*)carve((size_t)QQ * CC * EE * 2);
  _Float16* pospT = (_Float16*)carve((size_t)BB * CC * EE * 2);
  _Float16* ft    = (_Float16*)carve((size_t)BB * HWH * EE * 2);
  float*    s_b   = (float*)   carve((size_t)BB * HWH * CC * 4);
  float*    pos_b = (float*)   carve((size_t)BB * HWH * CC * 4);
  float*    neg0  = (float*)   carve((size_t)BB * HWH * CC * 4);
  float*    sums  = (float*)   carve((size_t)CC * 4);
  float*    cnts  = (float*)   carve((size_t)CC * 4);

  k_softmax_T<<<dim3((BB * HWH + 255) / 256), dim3(256), 0, stream>>>(score, smT);

  {
    size_t total = (size_t)(QQ - BB) * CC * EE;
    k_queue_t<<<dim3((unsigned)((total + 255) / 256)), dim3(256), 0, stream>>>(queue, qt);
  }

  k_protos<<<dim3(BB * CC), dim3(256), 0, stream>>>(feature, smT, center, qt, pospT);

  k_featn_t<<<dim3(BB * (HWH / 32)), dim3(256), 0, stream>>>(feature, ft);

  {
    dim3 grid(HWH / 128, CC / 16, BB);  // (8, 5, 2)
    size_t dyn_lds = (size_t)2 * NQ * LQTILE * sizeof(_Float16);  // 135,168 B
    k_main<<<grid, dim3(256), dyn_lds, stream>>>(ft, qt, pospT, s_b, pos_b, neg0);
  }

  k_zero<<<dim3(1), dim3(256), 0, stream>>>(sums, CC);
  k_zero<<<dim3(1), dim3(256), 0, stream>>>(cnts, CC);
  k_loss_pix<<<dim3((BB * HWH + 255) / 256), dim3(256), 0, stream>>>(
      s_b, pos_b, neg0, center, sums, cnts);

  k_final<<<dim3(1), dim3(128), 0, stream>>>(sums, cnts, out);

  (void)in_sizes; (void)n_in; (void)out_size; (void)ws_size;
}
